// DPGATlayer_19378892439973
// MI455X (gfx1250) — matrix operand order repaired and checked by reading, not compiled, not run on hardware
//
#include <hip/hip_runtime.h>
#include <hip/hip_bf16.h>

// ---------------------------------------------------------------- constants
constexpr int B = 4;
constexpr int N = 8192;
constexpr int D = 128;
constexpr int NPAD = 136;                 // LDS row stride (bf16 elems), 272B: conflict-friendly
constexpr float QSCALE = 0.011048543456039805f;  // 1/(8*sqrt(128)) folded into q

typedef __bf16 v16bf __attribute__((ext_vector_type(16)));
typedef __bf16 v8bf  __attribute__((ext_vector_type(8)));
typedef float  v8f   __attribute__((ext_vector_type(8)));
typedef int    v4i   __attribute__((vector_size(16)));   // matches builtin param type

#define AS_GLOBAL __attribute__((address_space(1)))
#define AS_LDS    __attribute__((address_space(3)))

static __device__ __forceinline__ v16bf cat16(v8bf lo, v8bf hi) {
    return __builtin_shufflevector(lo, hi, 0,1,2,3,4,5,6,7,8,9,10,11,12,13,14,15);
}
static __device__ __forceinline__ v8f wmma_bf16(v16bf a, v16bf b, v8f c) {
    // D(16x16 f32) = A(16x32 bf16) * B(32x16 bf16) + C
    return __builtin_amdgcn_wmma_f32_16x16x32_bf16(false, a, false, b, (short)0, c, false, false);
}
// A-fragment slice (16x32 bf16) for this lane from a row-major [*,128] bf16 row.
static __device__ __forceinline__ v16bf load_afrag(const __bf16* rowptr, int kk, int half) {
    v8bf lo = *(const v8bf*)(rowptr + kk * 32 + half * 16);
    v8bf hi = *(const v8bf*)(rowptr + kk * 32 + half * 16 + 8);
    return cat16(lo, hi);
}
// B-fragment (32x16 bf16): lane holds column l16, K contiguous per half.
static __device__ __forceinline__ v16bf load_bfrag(const __bf16* p) {
    v8bf lo = *(const v8bf*)(p);
    v8bf hi = *(const v8bf*)(p + 8);
    return cat16(lo, hi);
}
// One 16B-per-lane async global->LDS transfer (ASYNCcnt tracked).
static __device__ __forceinline__ void async_copy16(const __bf16* g, __bf16* l) {
    __builtin_amdgcn_global_load_async_to_lds_b128(
        (AS_GLOBAL v4i*)(AS_GLOBAL void*)(__bf16*)g,
        (AS_LDS v4i*)(AS_LDS void*)l, 0, 0);
}

// ---------------------------------------------------------------- prep: x -> bf16 (row-major + transposed)
__global__ void dpgat_cvt_x(const float* __restrict__ x,
                            __bf16* __restrict__ xh, __bf16* __restrict__ xhT) {
    const size_t total = (size_t)B * N * D;
    for (size_t i = (size_t)blockIdx.x * blockDim.x + threadIdx.x; i < total;
         i += (size_t)gridDim.x * blockDim.x) {
        int d = (int)(i % D);
        size_t bn = i / D;
        int nr = (int)(bn % N);
        int b = (int)(bn / N);
        float v = x[i];
        xh[i] = (__bf16)v;
        xhT[((size_t)b * D + d) * N + nr] = (__bf16)v;
    }
}

// ---------------------------------------------------------------- prep: weights -> bf16 transposed
__global__ void dpgat_cvt_w(const float* __restrict__ q, const float* __restrict__ k,
                            const float* __restrict__ v,
                            __bf16* __restrict__ qhT, __bf16* __restrict__ khT,
                            __bf16* __restrict__ vhT) {
    int i = blockIdx.x * blockDim.x + threadIdx.x;
    if (i < D * D) {
        int r = i / D, c = i % D;
        qhT[c * D + r] = (__bf16)(q[i] * QSCALE);   // fold (1/sqrt(D))/8 into Q
        khT[c * D + r] = (__bf16)(k[i]);
        vhT[c * D + r] = (__bf16)(v[i]);
    }
}

// ---------------------------------------------------------------- Q = x@q', K = x@k' (bf16 WMMA)
__global__ __launch_bounds__(128) void dpgat_proj(const __bf16* __restrict__ xh,
                                                  const __bf16* __restrict__ qhT,
                                                  const __bf16* __restrict__ khT,
                                                  __bf16* __restrict__ Qh,
                                                  __bf16* __restrict__ Kh) {
    const int tid = threadIdx.x, wave = tid >> 5, lane = tid & 31;
    const int half = lane >> 4, l16 = lane & 15;
    const int bpb = N / 64;
    const int b = blockIdx.x / bpb;
    const int row0 = (blockIdx.x % bpb) * 64 + wave * 16;

    v16bf xf[4];
    const __bf16* xr = xh + ((size_t)b * N + row0 + l16) * D;
#pragma unroll
    for (int kk = 0; kk < 4; ++kk) xf[kk] = load_afrag(xr, kk, half);

#pragma unroll
    for (int which = 0; which < 2; ++which) {
        const __bf16* wT = which ? khT : qhT;
        __bf16* dst = which ? Kh : Qh;
#pragma unroll
        for (int dt = 0; dt < 8; ++dt) {
            v8f acc = {};
#pragma unroll
            for (int kk = 0; kk < 4; ++kk)
                acc = wmma_bf16(xf[kk],
                                load_bfrag(wT + (dt * 16 + l16) * D + kk * 32 + half * 16), acc);
            __bf16* op = dst + ((size_t)b * N + row0 + half * 8) * D + dt * 16 + l16;
#pragma unroll
            for (int r = 0; r < 8; ++r) op[(size_t)r * D] = (__bf16)acc[r];
        }
    }
}

// ---------------------------------------------------------------- fused masked attention + output proj
__global__ __launch_bounds__(128) void dpgat_main(const __bf16* __restrict__ Qh,
                                                  const __bf16* __restrict__ Kh,
                                                  const __bf16* __restrict__ xhT,
                                                  const float* __restrict__ mask,
                                                  const __bf16* __restrict__ vhT,
                                                  float* __restrict__ out) {
    // double-buffered chunk tiles: [buf][ K(0) / xT(1) ][128 rows * NPAD]
    __shared__ __bf16 sBuf[2][2][128 * NPAD];
    __shared__ __bf16 sStage[4 * 16 * NPAD];   // per-wave C->A relayout buffer

    const int tid = threadIdx.x, wave = tid >> 5, lane = tid & 31;
    const int half = lane >> 4, l16 = lane & 15;
    const int bpb = N / 64;
    const int b = blockIdx.x / bpb;
    const int row0 = (blockIdx.x % bpb) * 64 + wave * 16;
    __bf16* stg = &sStage[wave * 16 * NPAD];

    const __bf16* kgB = Kh + (size_t)b * N * D;     // [key][d] row-major
    const __bf16* xgB = xhT + (size_t)b * D * N;    // [d][key] row-major

    // issue one chunk's async loads into buffer `buf` (16+16 instructions per wave)
    auto prefetch = [&](int chunk, int buf) {
        const __bf16* kg = kgB + (size_t)chunk * 128 * D;
        const __bf16* xg = xgB + chunk * 128;
        __bf16* dK = &sBuf[buf][0][0];
        __bf16* dX = &sBuf[buf][1][0];
        for (int qc = tid; qc < 2048; qc += 128) {
            int r = qc >> 4, c8 = qc & 15;
            async_copy16(kg + (size_t)r * D + c8 * 8, &dK[r * NPAD + c8 * 8]);
        }
        for (int qc = tid; qc < 2048; qc += 128) {
            int r = qc >> 4, c8 = qc & 15;
            async_copy16(xg + (size_t)r * N + c8 * 8, &dX[r * NPAD + c8 * 8]);
        }
    };

    // this wave's 16 query rows -> A fragments (folded scale makes S = A_ref/8)
    v16bf qf[4];
    {
        const __bf16* qr = Qh + ((size_t)b * N + row0 + l16) * D;
#pragma unroll
        for (int kk = 0; kk < 4; ++kk) qf[kk] = load_afrag(qr, kk, half);
    }

    v8f yacc[8];
#pragma unroll
    for (int dt = 0; dt < 8; ++dt) yacc[dt] = (v8f){};
    float rowsum[8] = {0.f, 0.f, 0.f, 0.f, 0.f, 0.f, 0.f, 0.f};

    prefetch(0, 0);

    for (int c = 0; c < N / 128; ++c) {
        const int cur = c & 1;
        const int nbase = c * 128;
        if (c + 1 < N / 128) {
            prefetch(c + 1, cur ^ 1);
            asm volatile("s_wait_asynccnt 0x20" ::: "memory");  // current chunk resident
        } else {
            asm volatile("s_wait_asynccnt 0x0" ::: "memory");
        }
        __syncthreads();                      // all waves' async data visible
        const __bf16* sK = &sBuf[cur][0][0];
        const __bf16* sX = &sBuf[cur][1][0];

#pragma unroll 1
        for (int cp = 0; cp < 4; ++cp) {
#pragma unroll
            for (int t = 0; t < 2; ++t) {
                const int ct = cp * 2 + t;
                v8f s = {};
#pragma unroll
                for (int kk = 0; kk < 4; ++kk)
                    s = wmma_bf16(qf[kk],
                                  load_bfrag(&sK[(ct * 16 + l16) * NPAD + kk * 32 + half * 16]), s);
                // softclip -> exp -> adjacency mask -> rowsum; stage P tile as bf16
                const float* mrow = mask + (size_t)(row0 + half * 8) * N + nbase + ct * 16 + l16;
#pragma unroll
                for (int r = 0; r < 8; ++r) {
                    float a = fminf(fmaxf(s[r], -15.f), 15.f);   // s == A_ref/8 already
                    float e2 = __expf(2.f * a);
                    float th = (e2 - 1.f) * __builtin_amdgcn_rcpf(e2 + 1.f);  // tanh
                    float p = __expf(8.f * th) * mrow[(size_t)r * N];
                    rowsum[r] += p;
                    stg[(half * 8 + r) * NPAD + t * 16 + l16] = (__bf16)p;
                }
            }
            asm volatile("s_wait_dscnt 0x0" ::: "memory");
            // P (16x32) A-fragment from stage, accumulate P @ x_chunk
            v16bf pf;
            {
                const __bf16* pr = &stg[l16 * NPAD];
                pf = cat16(*(const v8bf*)(pr + half * 16), *(const v8bf*)(pr + half * 16 + 8));
            }
#pragma unroll
            for (int dt = 0; dt < 8; ++dt)
                yacc[dt] = wmma_bf16(pf,
                                     load_bfrag(&sX[(dt * 16 + l16) * NPAD + cp * 32 + half * 16]),
                                     yacc[dt]);
        }
        __syncthreads();   // everyone done reading buf[cur] before it is refilled
    }

    // row-sum reduction within each 16-lane half (rows half*8 + r)
#pragma unroll
    for (int r = 0; r < 8; ++r) {
        float v = rowsum[r];
        v += __shfl_xor(v, 1, 32);
        v += __shfl_xor(v, 2, 32);
        v += __shfl_xor(v, 4, 32);
        v += __shfl_xor(v, 8, 32);
        rowsum[r] = 1.f / v;
    }

    // normalize, restage Y (16x128), apply output projection v (vT frags from L2)
#pragma unroll
    for (int dt = 0; dt < 8; ++dt)
#pragma unroll
        for (int r = 0; r < 8; ++r)
            stg[(half * 8 + r) * NPAD + dt * 16 + l16] = (__bf16)(yacc[dt][r] * rowsum[r]);
    asm volatile("s_wait_dscnt 0x0" ::: "memory");

    v16bf yf[4];
#pragma unroll
    for (int kk = 0; kk < 4; ++kk) yf[kk] = load_afrag(&stg[l16 * NPAD], kk, half);

#pragma unroll
    for (int dt = 0; dt < 8; ++dt) {
        v8f o = {};
#pragma unroll
        for (int kk = 0; kk < 4; ++kk)
            o = wmma_bf16(yf[kk],
                          load_bfrag(vhT + (dt * 16 + l16) * D + kk * 32 + half * 16), o);
        float* op = out + ((size_t)b * N + row0 + half * 8) * D + dt * 16 + l16;
#pragma unroll
        for (int r = 0; r < 8; ++r) op[(size_t)r * D] = o[r];
    }
}

// ---------------------------------------------------------------- launcher
extern "C" void kernel_launch(void* const* d_in, const int* in_sizes, int n_in,
                              void* d_out, int out_size, void* d_ws, size_t ws_size,
                              hipStream_t stream) {
    const float* x    = (const float*)d_in[0];
    const float* Am   = (const float*)d_in[1];
    const float* q    = (const float*)d_in[2];
    const float* k    = (const float*)d_in[3];
    const float* v    = (const float*)d_in[4];
    float* out        = (float*)d_out;

    char* ws = (char*)d_ws;
    const size_t SZ_XND = (size_t)B * N * D * sizeof(__bf16);   // 8 MiB
    const size_t SZ_W   = (size_t)D * D * sizeof(__bf16);       // 32 KiB
    __bf16* xh  = (__bf16*)(ws);
    __bf16* xhT = (__bf16*)(ws + SZ_XND);
    __bf16* Qh  = (__bf16*)(ws + 2 * SZ_XND);
    __bf16* Kh  = (__bf16*)(ws + 3 * SZ_XND);
    __bf16* qhT = (__bf16*)(ws + 4 * SZ_XND);
    __bf16* khT = (__bf16*)(ws + 4 * SZ_XND + SZ_W);
    __bf16* vhT = (__bf16*)(ws + 4 * SZ_XND + 2 * SZ_W);

    dpgat_cvt_x<<<2048, 256, 0, stream>>>(x, xh, xhT);
    dpgat_cvt_w<<<(D * D + 255) / 256, 256, 0, stream>>>(q, k, v, qhT, khT, vhT);
    dpgat_proj<<<B * N / 64, 128, 0, stream>>>(xh, qhT, khT, Qh, Kh);
    dpgat_main<<<B * N / 64, 128, 0, stream>>>(Qh, Kh, xhT, Am, vhT, out);
}
